// AAt_36953898614800
// MI455X (gfx1250) — matrix operand order repaired and checked by reading, not compiled, not run on hardware
//
#include <hip/hip_runtime.h>
#include <hip/hip_bf16.h>
#include <stdint.h>

typedef __attribute__((ext_vector_type(16))) _Float16 v16h;
typedef __attribute__((ext_vector_type(8)))  _Float16 v8h;
typedef __attribute__((ext_vector_type(8)))  float    v8f;

#define Bn   32
#define Cc   384
#define Hh   32
#define Ww   32
#define Nn   1024
#define Ssp  4
#define Nne  256
#define BeF  128
#define NH   6
#define HD   64

union U16 { uint4 u; v8h h; };

// Build a v16h fragment from two contiguous 16-byte chunks.
// A-fragment: chunks at p and p+16 halfs. B-fragment: chunks at p and p+16 halfs.
__device__ __forceinline__ v16h frag2(const _Float16* p0, const _Float16* p1) {
  U16 a, b;
  a.u = *reinterpret_cast<const uint4*>(p0);
  b.u = *reinterpret_cast<const uint4*>(p1);
  v16h r;
#pragma unroll
  for (int e = 0; e < 8; ++e) { r[e] = a.h[e]; r[8 + e] = b.h[e]; }
  return r;
}

// CDNA5 async global->LDS copy, 16B per lane (tracked by ASYNCcnt).
__device__ __forceinline__ void async_ld16(void* lds, const void* g) {
  uint32_t lofs = (uint32_t)(uintptr_t)lds;  // low 32 bits of flat LDS addr = LDS offset
  asm volatile("global_load_async_to_lds_b128 %0, %1, off"
               :: "v"(lofs), "v"(g) : "memory");
}
__device__ __forceinline__ void wait_async0() {
  asm volatile("s_wait_asynccnt 0" ::: "memory");
}

// ============ Kernel 0a: convert Wq|Wk|Wv|Wproj fp32 -> f16, concatenated ============
__global__ void wcvt_kernel(const float* __restrict__ Wq, const float* __restrict__ Wk,
                            const float* __restrict__ Wv, const float* __restrict__ Wp,
                            _Float16* __restrict__ W16) {
  int idx = blockIdx.x * 256 + threadIdx.x;     // over 4*C*C
  int which = idx / (Cc * Cc), r = idx % (Cc * Cc);
  const float* src = (which == 0) ? Wq : (which == 1) ? Wk : (which == 2) ? Wv : Wp;
  W16[idx] = (_Float16)src[r];
}

// ============ Kernel 0b: xT16[b][n][c] = (f16) x[b][c][n]  (LDS tile transpose) ============
__global__ void xT_kernel(const float* __restrict__ x, _Float16* __restrict__ xT16) {
  __shared__ _Float16 t[32][33];
  int b = blockIdx.z;
  int n0 = blockIdx.x * 32, c0 = blockIdx.y * 32;
  int tc = threadIdx.x & 31, tr = threadIdx.x >> 5;
#pragma unroll
  for (int it = 0; it < 4; ++it) {
    int c = tr + it * 8;
    t[c][tc] = (_Float16)x[((size_t)b * Cc + c0 + c) * Nn + n0 + tc];
  }
  __syncthreads();
#pragma unroll
  for (int it = 0; it < 4; ++it) {
    int n = tr + it * 8;
    xT16[((size_t)b * Nn + n0 + n) * Cc + c0 + tc] = t[tc][n];
  }
}

// ============ Kernel 1: fused QKV projection GEMM (64x256 macro-tile, LDS-staged A) ======
// Wave tile 2M x 4N (8 accumulators): A reused 4x, B reused 2x per k-step.
__global__ void qkv_kernel(const _Float16* __restrict__ W16,
                           const _Float16* __restrict__ xT16,
                           _Float16* __restrict__ qT16,   // [B*S][NH][Ne][HD]
                           _Float16* __restrict__ kT16,   // [B*S][NH][Ne][HD]
                           float* __restrict__ v32) {     // [B][C][N]
  __shared__ _Float16 Wl[64 * Cc];                        // 48KB weight panel
  int tid = threadIdx.x;
  int lane = tid & 31, wave = tid >> 5;
  int lo = lane & 15, hi = lane >> 4;
  int b = blockIdx.z / 3, which = blockIdx.z % 3;
  int crow0 = blockIdx.y * 64;                            // output-channel panel
  int col0 = blockIdx.x * 256;                            // pixel panel
  int mi = wave >> 2, ni = wave & 3;                      // wave: 32 M-rows x 64 N-cols

  const _Float16* Wsrc = W16 + (size_t)which * Cc * Cc + (size_t)crow0 * Cc;
  for (int i = tid; i < (64 * Cc) / 8; i += 256)
    async_ld16(&Wl[i * 8], Wsrc + (size_t)i * 8);
  wait_async0();
  __syncthreads();

  const _Float16* xb = xT16 + (size_t)b * Nn * Cc;
  v8f acc[2][4] = {};
  for (int kk = 0; kk < Cc; kk += 32) {
    v16h a[2], bf[4];
#pragma unroll
    for (int t = 0; t < 2; ++t) {
      const _Float16* pa = &Wl[(mi * 32 + t * 16 + lo) * Cc + kk + hi * 8];
      a[t] = frag2(pa, pa + 16);                          // ds_load_b128 x2
    }
#pragma unroll
    for (int t = 0; t < 4; ++t) {
      const _Float16* pb = xb + (size_t)(col0 + ni * 64 + t * 16 + lo) * Cc + kk + hi * 8;
      __builtin_prefetch(pb + 32, 0, 3);
      bf[t] = frag2(pb, pb + 16);                         // global_load_b128 x2
    }
#pragma unroll
    for (int tm = 0; tm < 2; ++tm)
#pragma unroll
      for (int tn = 0; tn < 4; ++tn)
        acc[tm][tn] = __builtin_amdgcn_wmma_f32_16x16x32_f16(
            false, a[tm], false, bf[tn], (short)0, acc[tm][tn], false, false);
  }

  if (which == 2) {                                       // V -> fp32 [b][c][nf] for PE conv
#pragma unroll
    for (int tm = 0; tm < 2; ++tm)
#pragma unroll
      for (int tn = 0; tn < 4; ++tn)
#pragma unroll
        for (int j = 0; j < 8; ++j) {
          int c = crow0 + mi * 32 + tm * 16 + hi * 8 + j;
          int nf = col0 + ni * 64 + tn * 16 + lo;
          v32[((size_t)b * Cc + c) * Nn + nf] = acc[tm][tn][j];
        }
  } else {                                                // Q/K -> f16 [be][h][n][d], d contig
    _Float16* dst = (which == 0) ? qT16 : kT16;
#pragma unroll
    for (int tm = 0; tm < 2; ++tm)
#pragma unroll
      for (int tn = 0; tn < 4; ++tn) {
        int c0 = crow0 + mi * 32 + tm * 16 + hi * 8;
        int nf = col0 + ni * 64 + tn * 16 + lo;
        int s = nf >> 8, n = nf & 255;
        int h = c0 / HD, d0 = c0 % HD;                    // j runs d0..d0+7 contiguously
        U16 pk;
#pragma unroll
        for (int j = 0; j < 8; ++j) pk.h[j] = (_Float16)acc[tm][tn][j];
        *reinterpret_cast<uint4*>(
            &dst[((((size_t)(b * Ssp + s) * NH + h) * Nne + n) * HD + d0)]) = pk.u;
      }
  }
}

// ============ Kernel 2: depthwise 3x3 PE, v' = v + dwconv(v), f16 out [b][c][nf] ============
__global__ void pe_kernel(const float* __restrict__ v32,
                          const float* __restrict__ Wpe,
                          _Float16* __restrict__ vpe16) {
  int idx = blockIdx.x * 256 + threadIdx.x;
  int w = idx & 31;
  int t = idx >> 5;
  int h = t & 31; t >>= 5;
  int c = t % Cc;
  int b = t / Cc;
  const float* vb = v32 + ((size_t)b * Cc + c) * Nn;
  float acc = vb[h * Ww + w];
#pragma unroll
  for (int dy = -1; dy <= 1; ++dy)
#pragma unroll
    for (int dx = -1; dx <= 1; ++dx) {
      int hh = h + dy, ww = w + dx;
      if (hh >= 0 && hh < Hh && ww >= 0 && ww < Ww)
        acc += Wpe[c * 9 + (dy + 1) * 3 + (dx + 1)] * vb[hh * Ww + ww];
    }
  vpe16[idx] = (_Float16)acc;
}

// ============ Kernel 3: attention, one block per (be, head, 64-row query block) ============
// LDS: K tile 32KB | V tile 32KB | scores 64KB | probs 32KB  = 160KB dynamic.
__global__ void attn_kernel(const _Float16* __restrict__ qT16,
                            const _Float16* __restrict__ kT16,
                            const _Float16* __restrict__ vpe16,
                            _Float16* __restrict__ avT16) {   // [b][nf][c]
  extern __shared__ char smem_raw[];
  _Float16* kl = (_Float16*)smem_raw;                         // [256][64]
  _Float16* vl = kl + Nne * HD;                               // [64][256]
  float*    sc = (float*)(vl + HD * Nne);                     // [64][256] fp32
  _Float16* pr = (_Float16*)(sc + 64 * 256);                  // [64][256] f16

  int tid = threadIdx.x;
  int lane = tid & 31, wave = tid >> 5;
  int lo = lane & 15, hi = lane >> 4;
  int qb = blockIdx.x, hh = blockIdx.y, be = blockIdx.z;
  int b = be >> 2, s = be & 3;
  int qrow0 = qb * 64;

  const _Float16* qbase = qT16 + ((size_t)be * NH + hh) * Nne * HD;
  const _Float16* kbase = kT16 + ((size_t)be * NH + hh) * Nne * HD;
  const _Float16* vbase = vpe16 + ((size_t)b * Cc + hh * HD) * Nn + s * Nne;

  // --- async-stage K [256][64] (contiguous) and V [64][256] (64 rows, stride Nn) ---
  for (int i = tid; i < (Nne * HD) / 8; i += 256)             // 2048 chunks of 16B
    async_ld16(&kl[i * 8], kbase + (size_t)i * 8);
  for (int i = tid; i < (HD * Nne) / 8; i += 256) {
    int row = i >> 5, ch = i & 31;                            // 32 chunks per 256-half row
    async_ld16(&vl[row * Nne + ch * 8], vbase + (size_t)row * Nn + ch * 8);
  }
  wait_async0();
  __syncthreads();

  // ---- scores: wave owns row-strip tm = wave>>1, 8 N-tiles; A reused 8x ----
  {
    int tm = wave >> 1, half = wave & 1;
    v8f acc[8] = {};
#pragma unroll
    for (int kk = 0; kk < HD; kk += 32) {
      const _Float16* pa = qbase + (size_t)(qrow0 + tm * 16 + lo) * HD + kk + hi * 8;
      v16h a = frag2(pa, pa + 16);
#pragma unroll
      for (int t = 0; t < 8; ++t) {
        const _Float16* pb = &kl[(half * 128 + t * 16 + lo) * HD + kk + hi * 8];
        v16h bf = frag2(pb, pb + 16);                     // ds_load_b128 x2
        acc[t] = __builtin_amdgcn_wmma_f32_16x16x32_f16(false, a, false, bf,
                                                        (short)0, acc[t], false, false);
      }
    }
#pragma unroll
    for (int t = 0; t < 8; ++t)
#pragma unroll
      for (int j = 0; j < 8; ++j)
        sc[(tm * 16 + hi * 8 + j) * 256 + half * 128 + t * 16 + lo] = acc[t][j] * 0.125f;
  }
  __syncthreads();

  // ---- row softmax, 4 lanes per row (wave32 shfl_xor groups of 4) ----
  {
    int r = tid >> 2, q = tid & 3;
    float* row = sc + r * 256;
    float mx = -1e30f;
    for (int i = q; i < 256; i += 4) mx = fmaxf(mx, row[i]);
    mx = fmaxf(mx, __shfl_xor(mx, 1, 32));
    mx = fmaxf(mx, __shfl_xor(mx, 2, 32));
    float ssum = 0.f;
    for (int i = q; i < 256; i += 4) { float e = __expf(row[i] - mx); row[i] = e; ssum += e; }
    ssum += __shfl_xor(ssum, 1, 32);
    ssum += __shfl_xor(ssum, 2, 32);
    float inv = 1.f / ssum;
    _Float16* prow = pr + r * 256;
    for (int i = q; i < 256; i += 4) prow[i] = (_Float16)(row[i] * inv);
  }
  __syncthreads();

  // ---- AV: wave owns col-strip td = wave>>1; B reused 2x; all operands in LDS ----
  {
    int td = wave >> 1, half = wave & 1;
    v8f acc[2] = {};
#pragma unroll
    for (int kk = 0; kk < 256; kk += 32) {
      const _Float16* pb = &vl[(td * 16 + lo) * 256 + kk + hi * 8];
      v16h bf = frag2(pb, pb + 16);
#pragma unroll
      for (int u = 0; u < 2; ++u) {
        int tm = half * 2 + u;
        const _Float16* pa = pr + (tm * 16 + lo) * 256 + kk + hi * 8;
        v16h a = frag2(pa, pa + 16);
        acc[u] = __builtin_amdgcn_wmma_f32_16x16x32_f16(false, a, false, bf,
                                                        (short)0, acc[u], false, false);
      }
    }
    // faithful torch reshape scramble: flat=(h*64+d)*256+n -> (n2,c2); nf=n2*4+s
#pragma unroll
    for (int u = 0; u < 2; ++u)
#pragma unroll
      for (int j = 0; j < 8; ++j) {
        int n = qrow0 + (half * 2 + u) * 16 + hi * 8 + j;
        int d = td * 16 + lo;
        int flat = (hh * HD + d) * Nne + n;
        int n2 = flat / Cc, c2 = flat % Cc;
        avT16[((size_t)b * Nn + n2 * Ssp + s) * Cc + c2] = (_Float16)acc[u][j];
      }
  }
}

// ============ Kernel 4: projection GEMM (64x256 macro-tile, fp32 out) ============
__global__ void proj_kernel(const _Float16* __restrict__ W16,     // quadrant 3 = Wproj
                            const _Float16* __restrict__ avT16,   // [b][nf][c]
                            float* __restrict__ po) {             // [b][c][nf]
  __shared__ _Float16 Wl[64 * Cc];
  int tid = threadIdx.x;
  int lane = tid & 31, wave = tid >> 5;
  int lo = lane & 15, hi = lane >> 4;
  int b = blockIdx.z;
  int crow0 = blockIdx.y * 64;
  int col0 = blockIdx.x * 256;
  int mi = wave >> 2, ni = wave & 3;

  const _Float16* Wsrc = W16 + (size_t)3 * Cc * Cc + (size_t)crow0 * Cc;
  for (int i = tid; i < (64 * Cc) / 8; i += 256)
    async_ld16(&Wl[i * 8], Wsrc + (size_t)i * 8);
  wait_async0();
  __syncthreads();

  const _Float16* inb = avT16 + (size_t)b * Nn * Cc;
  v8f acc[2][4] = {};
  for (int kk = 0; kk < Cc; kk += 32) {
    v16h a[2], bf[4];
#pragma unroll
    for (int t = 0; t < 2; ++t) {
      const _Float16* pa = &Wl[(mi * 32 + t * 16 + lo) * Cc + kk + hi * 8];
      a[t] = frag2(pa, pa + 16);
    }
#pragma unroll
    for (int t = 0; t < 4; ++t) {
      const _Float16* pb = inb + (size_t)(col0 + ni * 64 + t * 16 + lo) * Cc + kk + hi * 8;
      __builtin_prefetch(pb + 32, 0, 3);
      bf[t] = frag2(pb, pb + 16);
    }
#pragma unroll
    for (int tm = 0; tm < 2; ++tm)
#pragma unroll
      for (int tn = 0; tn < 4; ++tn)
        acc[tm][tn] = __builtin_amdgcn_wmma_f32_16x16x32_f16(
            false, a[tm], false, bf[tn], (short)0, acc[tm][tn], false, false);
  }
#pragma unroll
  for (int tm = 0; tm < 2; ++tm)
#pragma unroll
    for (int tn = 0; tn < 4; ++tn)
#pragma unroll
      for (int j = 0; j < 8; ++j) {
        int c = crow0 + mi * 32 + tm * 16 + hi * 8 + j;
        int nf = col0 + ni * 64 + tn * 16 + lo;
        po[((size_t)b * Cc + c) * Nn + nf] = acc[tm][tn][j];
      }
}

// ============ Kernel 5: BatchNorm statistics, one block per channel ============
__global__ void bn_stats_kernel(const float* __restrict__ po, float* __restrict__ stats) {
  __shared__ float s1[256], s2[256];
  int c = blockIdx.x;
  float sum = 0.f, sq = 0.f;
  for (int i = threadIdx.x; i < Bn * Nn; i += 256) {
    int b = i >> 10, nf = i & 1023;
    float v = po[((size_t)b * Cc + c) * Nn + nf];
    sum += v; sq += v * v;
  }
  s1[threadIdx.x] = sum; s2[threadIdx.x] = sq;
  __syncthreads();
  for (int off = 128; off > 0; off >>= 1) {
    if (threadIdx.x < off) {
      s1[threadIdx.x] += s1[threadIdx.x + off];
      s2[threadIdx.x] += s2[threadIdx.x + off];
    }
    __syncthreads();
  }
  if (threadIdx.x == 0) {
    float invM = 1.f / (float)(Bn * Nn);
    float mu = s1[0] * invM;
    float var = s2[0] * invM - mu * mu;
    stats[c] = mu;
    stats[Cc + c] = rsqrtf(var + 1e-5f);
  }
}

// ============ Kernel 6: normalize + residual ============
__global__ void bn_res_kernel(const float* __restrict__ x,
                              const float* __restrict__ po,
                              const float* __restrict__ stats,
                              const float* __restrict__ gamma,
                              const float* __restrict__ beta,
                              float* __restrict__ out) {
  int idx = blockIdx.x * 256 + threadIdx.x;
  int c = (idx >> 10) % Cc;
  out[idx] = x[idx] + gamma[c] * (po[idx] - stats[c]) * stats[Cc + c] + beta[c];
}

extern "C" void kernel_launch(void* const* d_in, const int* in_sizes, int n_in,
                              void* d_out, int out_size, void* d_ws, size_t ws_size,
                              hipStream_t stream) {
  (void)in_sizes; (void)n_in; (void)out_size; (void)ws_size;
  const float* x     = (const float*)d_in[0];
  const float* Wq    = (const float*)d_in[1];
  const float* Wk    = (const float*)d_in[2];
  const float* Wv    = (const float*)d_in[3];
  const float* Wpe   = (const float*)d_in[4];
  const float* Wproj = (const float*)d_in[5];
  const float* gamma = (const float*)d_in[6];
  const float* beta  = (const float*)d_in[7];
  float* out = (float*)d_out;

  const size_t E = (size_t)Bn * Cc * Nn;     // 12.58M elements
  char* ws = (char*)d_ws;
  _Float16* W16   = (_Float16*)ws;           // 4*C*C
  _Float16* xT16  = W16 + (size_t)4 * Cc * Cc;
  _Float16* qT16  = xT16 + E;
  _Float16* kT16  = qT16 + E;
  _Float16* vpe16 = kT16 + E;
  _Float16* avT16 = vpe16 + E;
  float*    v32   = (float*)(avT16 + E);
  float*    po    = v32 + E;
  float*    stats = po + E;

  wcvt_kernel<<<dim3((4 * Cc * Cc) / 256), 256, 0, stream>>>(Wq, Wk, Wv, Wproj, W16);
  xT_kernel<<<dim3(Nn / 32, Cc / 32, Bn), 256, 0, stream>>>(x, xT16);
  qkv_kernel<<<dim3(Nn / 256, Cc / 64, Bn * 3), 256, 0, stream>>>(W16, xT16, qT16, kT16, v32);
  pe_kernel<<<dim3((unsigned)(E / 256)), 256, 0, stream>>>(v32, Wpe, vpe16);
  attn_kernel<<<dim3(4, NH, BeF), 256,
                (Nne * HD + HD * Nne) * 2 + 64 * 256 * 4 + 64 * 256 * 2, stream>>>(
      qT16, kT16, vpe16, avT16);
  proj_kernel<<<dim3(Nn / 256, Cc / 64, Bn), 256, 0, stream>>>(W16, avT16, po);
  bn_stats_kernel<<<dim3(Cc), 256, 0, stream>>>(po, stats);
  bn_res_kernel<<<dim3((unsigned)(E / 256)), 256, 0, stream>>>(x, po, stats, gamma, beta, out);
}
